// WinogradConv2dF23_56891136803398
// MI455X (gfx1250) — compile-verified, matrix operand order checked by reading, not run on hardware
//
#include <hip/hip_runtime.h>
#include <hip/hip_bf16.h>

typedef __attribute__((ext_vector_type(16))) __bf16 v16bf;
typedef __attribute__((ext_vector_type(8)))  float  v8f;

#define CIN   128
#define COUT  128
#define HW    112
#define NT    56                     // tiles per spatial dim
#define TPI   (NT*NT)                // 3136 tiles per image
#define TB    32                     // tiles per workgroup (2 WMMA n-tiles)
#define KC    32                     // channel chunk = bf16 WMMA K depth
#define TBLK  (TPI/TB)               // 98 tile-blocks per image

// A/B 16-bit fragment mapping (ISA 7.12.2, 16-bit A 16x32):
//   lane = (row&15) + 16*((k>>3)&1) ; elem e = (k&7) + 8*(k>>4)
__device__ __forceinline__ int frag_lane(int row, int k) { return (row & 15) + 16 * ((k >> 3) & 1); }
__device__ __forceinline__ int frag_elem(int k)          { return (k & 7) + 8 * (k >> 4); }

// ufrag layout: [p(16)][otile(8)][cchunk(4)][lane(32)][e(16)] bf16  -> 512 KB
#define UF_PSTRIDE (8*4*32*16)

// ---------------- Kernel 1: weight transform U = G w G^T, packed as A-fragments ----------------
__global__ __launch_bounds__(256) void wino_weight_xform(
    const float* __restrict__ w, __bf16* __restrict__ ufrag) {
  int idx = blockIdx.x * blockDim.x + threadIdx.x;   // 0..16383
  int o = idx >> 7;
  int c = idx & 127;
  const float* wp = w + (size_t)(o * CIN + c) * 9;
  float a00=wp[0],a01=wp[1],a02=wp[2],a10=wp[3],a11=wp[4],a12=wp[5],a20=wp[6],a21=wp[7],a22=wp[8];
  // g = G @ w  (4x3):  G = [[1,0,0],[.5,.5,.5],[.5,-.5,.5],[0,0,1]]
  float g[4][3];
  g[0][0]=a00;                 g[0][1]=a01;                 g[0][2]=a02;
  g[1][0]=0.5f*(a00+a10+a20);  g[1][1]=0.5f*(a01+a11+a21);  g[1][2]=0.5f*(a02+a12+a22);
  g[2][0]=0.5f*(a00-a10+a20);  g[2][1]=0.5f*(a01-a11+a21);  g[2][2]=0.5f*(a02-a12+a22);
  g[3][0]=a20;                 g[3][1]=a21;                 g[3][2]=a22;
  // U = g @ G^T (4x4)
  float U[4][4];
#pragma unroll
  for (int i = 0; i < 4; ++i) {
    U[i][0] = g[i][0];
    U[i][1] = 0.5f * (g[i][0] + g[i][1] + g[i][2]);
    U[i][2] = 0.5f * (g[i][0] - g[i][1] + g[i][2]);
    U[i][3] = g[i][2];
  }
  int ot = o >> 4, m = o & 15;
  int cc = c >> 5, kk = c & 31;
  int ln = frag_lane(m, kk);
  int e  = frag_elem(kk);
#pragma unroll
  for (int p = 0; p < 16; ++p)
    ufrag[(size_t)p * UF_PSTRIDE + (((ot * 4 + cc) * 32 + ln) * 16 + e)] = (__bf16)U[p >> 2][p & 3];
}

// ---------------- Kernel 2: fused input-xform -> 16x WMMA GEMM -> output-xform ----------------
__global__ __launch_bounds__(512, 1) void wino_main(
    const float* __restrict__ x, const __bf16* __restrict__ ufrag,
    const float* __restrict__ bias, float* __restrict__ y) {
  // V chunk in B-fragment order: [ntile(2)][p(16)][lane(32)][e(16)] bf16 = 32 KB
  __shared__ __align__(32) __bf16 ldsV[2][16][32][16];

  const int tid   = threadIdx.x;
  const int lane  = tid & 31;
  const int wave  = tid >> 5;        // 0..15
  const int otile = wave & 7;        // o-row block of 16
  const int wnt   = wave >> 3;       // which 16-tile slab of the 32
  const int n     = blockIdx.x / TBLK;
  const int tbase = (blockIdx.x % TBLK) * TB;

  v8f acc[16];
#pragma unroll
  for (int p = 0; p < 16; ++p) acc[p] = (v8f)(0.0f);

  const __bf16* ubase0 = ufrag + (size_t)((otile * 4) * 32 + lane) * 16;

  for (int c0 = 0; c0 < CIN; c0 += KC) {
    const int cc = c0 >> 5;
    // warm L2/L0 for this chunk's U fragments while we do the input transform
    __builtin_prefetch(ubase0 + (size_t)cc * (32 * 16), 0, 0);
    __syncthreads();
    // ---- produce V chunk: 32 channels x 32 tiles, 2 (kk,tile) pairs per thread ----
#pragma unroll
    for (int pi = tid; pi < KC * TB; pi += 512) {
      const int tile = pi & 31;
      const int kk   = pi >> 5;
      const int t    = tbase + tile;
      const int th = t / NT, tw = t % NT;
      const int h0 = 2 * th - 1, w0 = 2 * tw - 1;
      const float* xp = x + ((size_t)(n * CIN + (c0 + kk))) * (HW * HW);
      float d[4][4];
#pragma unroll
      for (int r = 0; r < 4; ++r) {
        const int h = h0 + r;
        const bool hv = (unsigned)h < (unsigned)HW;
        const float* row = xp + h * HW;
#pragma unroll
        for (int s = 0; s < 4; ++s) {
          const int ww = w0 + s;
          d[r][s] = (hv && (unsigned)ww < (unsigned)HW) ? row[ww] : 0.0f;
        }
      }
      // V = Bt d Bt^T
      float tr[4][4];
#pragma unroll
      for (int j = 0; j < 4; ++j) {
        tr[0][j] = d[0][j] - d[2][j];
        tr[1][j] = d[1][j] + d[2][j];
        tr[2][j] = d[2][j] - d[1][j];
        tr[3][j] = d[1][j] - d[3][j];
      }
      const int vl  = frag_lane(tile, kk);
      const int ve  = frag_elem(kk);
      const int vnt = tile >> 4;
#pragma unroll
      for (int i = 0; i < 4; ++i) {
        float v0 = tr[i][0] - tr[i][2];
        float v1 = tr[i][1] + tr[i][2];
        float v2 = tr[i][2] - tr[i][1];
        float v3 = tr[i][1] - tr[i][3];
        ldsV[vnt][i * 4 + 0][vl][ve] = (__bf16)v0;
        ldsV[vnt][i * 4 + 1][vl][ve] = (__bf16)v1;
        ldsV[vnt][i * 4 + 2][vl][ve] = (__bf16)v2;
        ldsV[vnt][i * 4 + 3][vl][ve] = (__bf16)v3;
      }
    }
    __syncthreads();
    // ---- consume: one WMMA per frequency point, f32 accumulate ----
    const __bf16* ub = ubase0 + (size_t)cc * (32 * 16);
#pragma unroll
    for (int p = 0; p < 16; ++p) {
      v16bf a = *(const v16bf*)(ub + (size_t)p * UF_PSTRIDE);
      v16bf b = *(const v16bf*)(&ldsV[wnt][p][lane][0]);
      acc[p] = __builtin_amdgcn_wmma_f32_16x16x32_bf16(
          false, a, false, b, (short)0, acc[p], false, false);
    }
  }

  // ---- epilogue: per-lane A^T M A + bias; each lane owns 8 (o,tile) pairs, all 16 points ----
  const int ncol = lane & 15;
  const int mhi  = lane >> 4;                // +8 on o-row for upper half-lanes
  const int t    = tbase + wnt * 16 + ncol;
  const int th = t / NT, tw = t % NT;
#pragma unroll
  for (int r = 0; r < 8; ++r) {
    const int o = otile * 16 + r + 8 * mhi;
    float m[16];
#pragma unroll
    for (int p = 0; p < 16; ++p) m[p] = acc[p][r];
    float z[2][4];
#pragma unroll
    for (int j = 0; j < 4; ++j) {
      z[0][j] = m[j] + m[4 + j] + m[8 + j];
      z[1][j] = m[4 + j] - m[8 + j] - m[12 + j];
    }
    const float bv = bias[o];
    float* yo = y + ((size_t)(n * COUT + o)) * (HW * HW) + (2 * th) * HW + 2 * tw;
    yo[0]      = z[0][0] + z[0][1] + z[0][2] + bv;
    yo[1]      = z[0][1] - z[0][2] - z[0][3] + bv;
    yo[HW]     = z[1][0] + z[1][1] + z[1][2] + bv;
    yo[HW + 1] = z[1][1] - z[1][2] - z[1][3] + bv;
  }
}

extern "C" void kernel_launch(void* const* d_in, const int* in_sizes, int n_in,
                              void* d_out, int out_size, void* d_ws, size_t ws_size,
                              hipStream_t stream) {
  (void)in_sizes; (void)n_in; (void)out_size; (void)ws_size;
  const float* x = (const float*)d_in[0];
  const float* w = (const float*)d_in[1];
  const float* b = (const float*)d_in[2];
  float* y = (float*)d_out;
  __bf16* ufrag = (__bf16*)d_ws;   // 16*8*4*32*16 bf16 = 512 KB

  wino_weight_xform<<<64, 256, 0, stream>>>(w, ufrag);
  wino_main<<<16 * TBLK, 512, 0, stream>>>(x, ufrag, b, y);
}
